// GATv2JK_49280454754588
// MI455X (gfx1250) — matrix operand order repaired and checked by reading, not compiled, not run on hardware
//
#include <hip/hip_runtime.h>
#include <hip/hip_bf16.h>
#include <math.h>

// ---------------- static model config ----------------
enum {
  kB = 16, kNPG = 1024, kN = 16384, kE = 131072,
  kINC = 128, kHID = 256, kHEADS = 4, kL = 4,
  kEDIM = 16, kOUTC = 32, kC2 = 1024
};

// ---------------- WMMA types ----------------
typedef __attribute__((ext_vector_type(16))) _Float16 v16h;
typedef __attribute__((ext_vector_type(8)))  _Float16 v8h;
typedef __attribute__((ext_vector_type(8)))  float    v8f;

union HFrag { v16h v; v8h h2[2]; _Float16 e[16]; };
union CFrag { v8f  v; float e[8]; };
union HPack { _Float16 h[2]; unsigned u; };

// =====================================================================
// Generic WMMA GEMM:  C[M,N] = act( A[M,K] @ B[K,N] + bias[N] + Cin[M,N] )
// Block: 256 threads = 8 wave32 arranged 2(M) x 4(N). Each wave computes a
// 32x32 tile as 2x2 v_wmma_f32_16x16x32_f16; block tile 64x128, BK=32.
// A staged row-major in LDS (fragment = 2 contiguous b128 chunks/lane);
// B staged TRANSPOSED in LDS (fragment = 2 b128 chunks of 8 consecutive K).
// Bounds handled by clamp+mask (branch-free staging). Requires K % 32 == 0.
// =====================================================================
__global__ __launch_bounds__(256) void wmma_gemm_kernel(
    const float* __restrict__ A, const float* __restrict__ Bm,
    const float* __restrict__ bias, const float* __restrict__ Cin,
    float* __restrict__ C, int M, int N, int K, int relu)
{
  __shared__ _Float16 sA [64][40];   // 64 rows x 32 k (stride 40 halves = 80B)
  __shared__ _Float16 sBT[128][40];  // 128 cols x 32 k, transposed

  const int tid  = threadIdx.x;
  const int wave = tid >> 5;
  const int lane = tid & 31;
  const int wm   = wave >> 2;        // 0..1  (M direction)
  const int wn   = wave & 3;         // 0..3  (N direction)
  const int m0   = blockIdx.y * 64;
  const int n0   = blockIdx.x * 128;
  const int mrow = lane & 15;        // M row within 16 (N col for B/C frags)
  const int hi   = lane >> 4;        // lane-half selector

  CFrag c[2][2];
#pragma unroll
  for (int i = 0; i < 2; ++i)
#pragma unroll
    for (int j = 0; j < 2; ++j)
#pragma unroll
      for (int v = 0; v < 8; ++v) c[i][j].e[v] = 0.0f;

  for (int k0 = 0; k0 < K; k0 += 32) {
    // ---- stage A tile: 64 rows x 32 k, packed pair stores ----
#pragma unroll
    for (int t = 0; t < 4; ++t) {
      int p  = tid + t * 256;                 // 0..1023 pair index
      int r  = p >> 4;                        // 0..63
      int cc = (p & 15) * 2;                  // 0,2,..,30
      int gr = m0 + r;
      int grc = (gr < M) ? gr : (M - 1);
      float sel = (gr < M) ? 1.0f : 0.0f;
      const float2 v2 = *(const float2*)(A + (size_t)grc * K + k0 + cc);
      HPack pk;
      pk.h[0] = (_Float16)(v2.x * sel);
      pk.h[1] = (_Float16)(v2.y * sel);
      *(unsigned*)&sA[r][cc] = pk.u;
    }
    // ---- stage B tile transposed: 128 cols x 32 k ----
#pragma unroll
    for (int t = 0; t < 8; ++t) {
      int p  = tid + t * 256;                 // 0..2047 pair index
      int cc = p & 127;                       // column in tile
      int r  = (p >> 7) * 2;                  // 0,2,..,30 (k)
      int gc = n0 + cc;
      int gcc = (gc < N) ? gc : (N - 1);
      float sel = (gc < N) ? 1.0f : 0.0f;
      float b0 = Bm[(size_t)(k0 + r) * N + gcc];
      float b1 = Bm[(size_t)(k0 + r + 1) * N + gcc];
      HPack pk;
      pk.h[0] = (_Float16)(b0 * sel);
      pk.h[1] = (_Float16)(b1 * sel);
      *(unsigned*)&sBT[cc][r] = pk.u;
    }
    __syncthreads();

    // ---- fragments: contiguous b128 chunks per ISA wave32 layouts ----
    HFrag a[2], b[2];
#pragma unroll
    for (int sm = 0; sm < 2; ++sm) {
      int rr = wm * 32 + sm * 16 + mrow;
      a[sm].h2[0] = *(const v8h*)&sA[rr][hi * 8];        // k = hi*8 .. +7
      a[sm].h2[1] = *(const v8h*)&sA[rr][16 + hi * 8];   // k = 16+hi*8 .. +7
    }
#pragma unroll
    for (int sn = 0; sn < 2; ++sn) {
      int ccf = wn * 32 + sn * 16 + mrow;
      b[sn].h2[0] = *(const v8h*)&sBT[ccf][hi * 8];      // k = hi*8 .. +7
      b[sn].h2[1] = *(const v8h*)&sBT[ccf][16 + hi * 8]; // k = 16+hi*8 .. +7
    }
#pragma unroll
    for (int sm = 0; sm < 2; ++sm)
#pragma unroll
      for (int sn = 0; sn < 2; ++sn)
        c[sm][sn].v = __builtin_amdgcn_wmma_f32_16x16x32_f16(
            false, a[sm].v, false, b[sn].v, (short)0, c[sm][sn].v, false, false);
    __syncthreads();
  }

  // ---- epilogue: bias / residual-add / relu / store ----
#pragma unroll
  for (int sm = 0; sm < 2; ++sm) {
#pragma unroll
    for (int sn = 0; sn < 2; ++sn) {
      int col = n0 + wn * 32 + sn * 16 + mrow;
#pragma unroll
      for (int v = 0; v < 8; ++v) {
        int row = m0 + wm * 32 + sm * 16 + hi * 8 + v;
        if (row < M && col < N) {
          float val = c[sm][sn].e[v];
          if (bias) val += bias[col];
          if (Cin)  val += Cin[(size_t)row * N + col];
          if (relu) val = fmaxf(val, 0.0f);
          C[(size_t)row * N + col] = val;
        }
      }
    }
  }
}

// =====================================================================
// PyG LayerNorm(mode='graph'): normalize per graph over nodes AND channels.
// =====================================================================
__global__ __launch_bounds__(256) void graph_ln_kernel(
    const float* __restrict__ x, float* __restrict__ out,
    const float* __restrict__ w, const float* __restrict__ b, int C)
{
  __shared__ float s1[256], s2[256];
  const int tid = threadIdx.x;
  const size_t base  = (size_t)blockIdx.x * kNPG * C;
  const size_t total = (size_t)kNPG * C;
  float sum = 0.f, sq = 0.f;
  for (size_t i = tid; i < total; i += 256) {
    float v = x[base + i]; sum += v; sq += v * v;
  }
  s1[tid] = sum; s2[tid] = sq; __syncthreads();
  for (int off = 128; off > 0; off >>= 1) {
    if (tid < off) { s1[tid] += s1[tid + off]; s2[tid] += s2[tid + off]; }
    __syncthreads();
  }
  float mean = s1[0] / (float)total;
  float var  = s2[0] / (float)total - mean * mean;
  float rstd = rsqrtf(var + 1e-5f);
  for (size_t i = tid; i < total; i += 256) {
    int cc = (int)(i % C);
    out[base + i] = (x[base + i] - mean) * rstd * w[cc] + b[cc];
  }
}

// Row LayerNorm over last dim C, in place. One block per row.
__global__ __launch_bounds__(256) void row_ln_kernel(
    float* __restrict__ x, const float* __restrict__ w,
    const float* __restrict__ b, int C)
{
  __shared__ float s1[256], s2[256];
  const int tid = threadIdx.x;
  float* p = x + (size_t)blockIdx.x * C;
  float sum = 0.f, sq = 0.f;
  for (int i = tid; i < C; i += 256) { float v = p[i]; sum += v; sq += v * v; }
  s1[tid] = sum; s2[tid] = sq; __syncthreads();
  for (int off = 128; off > 0; off >>= 1) {
    if (tid < off) { s1[tid] += s1[tid + off]; s2[tid] += s2[tid + off]; }
    __syncthreads();
  }
  float mean = s1[0] / (float)C;
  float var  = s2[0] / (float)C - mean * mean;
  float rstd = rsqrtf(var + 1e-5f);
  for (int i = tid; i < C; i += 256) p[i] = (p[i] - mean) * rstd * w[i] + b[i];
}

// =====================================================================
// Edge passes for GATv2 (segment softmax over dst nodes).
// =====================================================================
__device__ inline void atomicMaxF(float* addr, float value) {
  float old = *addr;
  while (old < value) {
    int assumed = __float_as_int(old);
    int r = atomicCAS((int*)addr, assumed, __float_as_int(value));
    if (r == assumed) break;
    old = __int_as_float(r);
  }
}

// Pass 1: logit[e,h] = sum_c att[h,c]*leaky( xl[src]+xr[dst]+(edge_attr@we) )
__global__ __launch_bounds__(256) void edge_logit_kernel(
    const float* __restrict__ xl, const float* __restrict__ xr,
    const float* __restrict__ eattr, const float* __restrict__ we,
    const float* __restrict__ att, const int* __restrict__ src,
    const int* __restrict__ dst, float* __restrict__ logit,
    float* __restrict__ lmax, int OC)
{
  int t = blockIdx.x * 256 + threadIdx.x;
  if (t >= kE * kHEADS) return;
  int e = t / kHEADS, h = t % kHEADS;
  int s = src[e], d = dst[e];
  int hc = kHEADS * OC;
  float ea[kEDIM];
#pragma unroll
  for (int i = 0; i < kEDIM; ++i) ea[i] = eattr[(size_t)e * kEDIM + i];
  const float* xls = xl + (size_t)s * hc + h * OC;
  const float* xrd = xr + (size_t)d * hc + h * OC;
  const float* av  = att + h * OC;
  float acc = 0.f;
  for (int c = 0; c < OC; ++c) {
    float ev = 0.f;
#pragma unroll
    for (int i = 0; i < kEDIM; ++i) ev += ea[i] * we[(size_t)i * hc + h * OC + c];
    float m = xls[c] + xrd[c] + ev;
    m = (m > 0.f) ? m : m * 0.2f;        // leaky_relu(0.2)
    acc += m * av[c];
  }
  logit[t] = acc;
  atomicMaxF(&lmax[(size_t)d * kHEADS + h], acc);
}

// Pass 2: ex = exp(logit - lmax[dst]); den[dst] += ex
__global__ __launch_bounds__(256) void edge_exp_kernel(
    float* __restrict__ logit, const float* __restrict__ lmax,
    float* __restrict__ den, const int* __restrict__ dst)
{
  int t = blockIdx.x * 256 + threadIdx.x;
  if (t >= kE * kHEADS) return;
  int e = t / kHEADS, h = t % kHEADS;
  int d = dst[e];
  float ex = expf(logit[t] - lmax[(size_t)d * kHEADS + h]);
  logit[t] = ex;
  atomicAdd(&den[(size_t)d * kHEADS + h], ex);
}

// Pass 3: agg[dst,h,c] += xl[src,h,c] * ex/(den[dst,h]+1e-16)
__global__ __launch_bounds__(256) void edge_scatter_kernel(
    const float* __restrict__ ex, const float* __restrict__ den,
    const float* __restrict__ xl, float* __restrict__ agg,
    const int* __restrict__ src, const int* __restrict__ dst, int OC)
{
  int hc = kHEADS * OC;
  long long t = (long long)blockIdx.x * 256 + threadIdx.x;
  if (t >= (long long)kE * hc) return;
  int e   = (int)(t / hc);
  int rem = (int)(t % hc);
  int h   = rem / OC;
  int s = src[e], d = dst[e];
  float alpha = ex[(size_t)e * kHEADS + h] /
                (den[(size_t)d * kHEADS + h] + 1e-16f);
  atomicAdd(&agg[(size_t)d * hc + rem], xl[(size_t)s * hc + rem] * alpha);
}

// =====================================================================
// Small multi-head attention (softmax over keys). One block per (b,h,q).
// q,k,v: [B, L, C] row-major with C = H*HD. LK <= 1024, HD <= 256.
// =====================================================================
__global__ __launch_bounds__(256) void attn_kernel(
    const float* __restrict__ q, const float* __restrict__ k,
    const float* __restrict__ v, float* __restrict__ o,
    int H, int LQ, int LK, int HD)
{
  __shared__ float sc[1024];
  __shared__ float qs[256];
  __shared__ float red[256];
  const int tid = threadIdx.x;
  const int idx = blockIdx.x;
  const int qi = idx % LQ;
  const int h  = (idx / LQ) % H;
  const int b  = idx / (LQ * H);
  const int C  = H * HD;
  const float* qrow = q + ((size_t)(b * LQ + qi)) * C + h * HD;
  const float* kb   = k + (size_t)b * LK * C + h * HD;
  const float* vb   = v + (size_t)b * LK * C + h * HD;
  for (int d = tid; d < HD; d += 256) qs[d] = qrow[d];
  __syncthreads();
  const float scale = rsqrtf((float)HD);
  for (int kk = tid; kk < LK; kk += 256) {
    const float* kr = kb + (size_t)kk * C;
    float acc = 0.f;
    for (int d = 0; d < HD; ++d) acc += qs[d] * kr[d];
    sc[kk] = acc * scale;
  }
  __syncthreads();
  float m = -3.4e38f;
  for (int kk = tid; kk < LK; kk += 256) m = fmaxf(m, sc[kk]);
  red[tid] = m; __syncthreads();
  for (int off = 128; off > 0; off >>= 1) {
    if (tid < off) red[tid] = fmaxf(red[tid], red[tid + off]);
    __syncthreads();
  }
  m = red[0]; __syncthreads();
  float s = 0.f;
  for (int kk = tid; kk < LK; kk += 256) { float e2 = expf(sc[kk] - m); sc[kk] = e2; s += e2; }
  red[tid] = s; __syncthreads();
  for (int off = 128; off > 0; off >>= 1) {
    if (tid < off) red[tid] += red[tid + off];
    __syncthreads();
  }
  const float inv = 1.f / red[0];
  __syncthreads();
  for (int d = tid; d < HD; d += 256) {
    float acc = 0.f;
    for (int kk = 0; kk < LK; ++kk) acc += sc[kk] * vb[(size_t)kk * C + d];
    o[((size_t)(b * LQ + qi)) * C + h * HD + d] = acc * inv;
  }
}

// =====================================================================
// Elementwise helpers
// =====================================================================
__global__ void fill_kernel(float* p, float v, long long n) {
  long long i = (long long)blockIdx.x * 256 + threadIdx.x;
  if (i < n) p[i] = v;
}
__global__ void add_kernel(float* d, const float* s, long long n) {
  long long i = (long long)blockIdx.x * 256 + threadIdx.x;
  if (i < n) d[i] += s[i];
}
__global__ void bcast_kernel(float* d, const float* s, long long period, long long n) {
  long long i = (long long)blockIdx.x * 256 + threadIdx.x;
  if (i < n) d[i] = s[i % period];
}
__global__ void copy_slice_kernel(const float* s, float* d, int Cs, int Cd, int off, long long n) {
  long long t = (long long)blockIdx.x * 256 + threadIdx.x;
  if (t >= n) return;
  long long r = t / Cs; int c = (int)(t % Cs);
  d[r * Cd + off + c] = s[t];
}
__global__ void head_mean_kernel(const float* a, float* o, int OC, long long n) {
  long long t = (long long)blockIdx.x * 256 + threadIdx.x;
  if (t >= n) return;
  long long r = t / OC; int c = (int)(t % OC);
  float s = 0.f;
  for (int h = 0; h < kHEADS; ++h) s += a[r * (kHEADS * OC) + h * OC + c];
  o[t] = s * (1.0f / kHEADS);
}

// =====================================================================
// Host-side helpers
// =====================================================================
static inline void gemm(const float* A, const float* Bm, const float* bias,
                        const float* Cin, float* C, int M, int N, int K,
                        int relu, hipStream_t st) {
  dim3 g((N + 127) / 128, (M + 63) / 64);
  wmma_gemm_kernel<<<g, 256, 0, st>>>(A, Bm, bias, Cin, C, M, N, K, relu);
}
static inline void fill(float* p, float v, long long n, hipStream_t st) {
  fill_kernel<<<(unsigned)((n + 255) / 256), 256, 0, st>>>(p, v, n);
}

struct MabP {
  const float *bk, *bo, *bq, *bv, *lin_b, *lin_w, *ln1_b, *ln1_w,
              *ln2_b, *ln2_w, *wk, *wo, *wq, *wv;
};

// mab(x=qin, y=kv): mha + residual + ln1 + ffn + ln2  (C=1024, H=4, HD=256)
static void run_mab(const float* qin, const float* kv, float* z,
                    int Bb, int LQ, int LK, const MabP& m,
                    float* qt, float* kt, float* vt, float* ot, float* tt,
                    hipStream_t st) {
  const int rows = Bb * LQ, krows = Bb * LK;
  gemm(qin, m.wq, m.bq, nullptr, qt, rows,  kC2, kC2, 0, st);
  gemm(kv,  m.wk, m.bk, nullptr, kt, krows, kC2, kC2, 0, st);
  gemm(kv,  m.wv, m.bv, nullptr, vt, krows, kC2, kC2, 0, st);
  attn_kernel<<<Bb * kHEADS * LQ, 256, 0, st>>>(qt, kt, vt, ot, kHEADS, LQ, LK, kC2 / kHEADS);
  gemm(ot, m.wo, m.bo, /*Cin=+q_in*/ qin, z, rows, kC2, kC2, 0, st);
  row_ln_kernel<<<rows, 256, 0, st>>>(z, m.ln1_w, m.ln1_b, kC2);
  gemm(z, m.lin_w, m.lin_b, nullptr, tt, rows, kC2, kC2, /*relu*/1, st);
  long long nel = (long long)rows * kC2;
  add_kernel<<<(unsigned)((nel + 255) / 256), 256, 0, st>>>(z, tt, nel);
  row_ln_kernel<<<rows, 256, 0, st>>>(z, m.ln2_w, m.ln2_b, kC2);
}

extern "C" void kernel_launch(void* const* d_in, const int* in_sizes, int n_in,
                              void* d_out, int out_size, void* d_ws, size_t ws_size,
                              hipStream_t stream) {
  // ---- input pointer map (top-level dict order; params pytree flattened in
  //      JAX-canonical sorted-key order: gmt < layers < out < proj) ----
  auto F = [&](int i) { return (const float*)d_in[i]; };
  // 0:x 1:edge_attr
  // gmt.pma1: 2:lin_b 3:lin_w 4..17:mab(bk,bo,bq,bv,lin_b,lin_w,ln1_b,ln1_w,
  //                                     ln2_b,ln2_w,wk,wo,wq,wv) 18:seed
  // gmt.pma2: 19:lin_b 20:lin_w 21..34:mab 35:seed
  // gmt.sab : 36..49:mab
  // layers i: 50+10i: att,bias,bl,br,ln_b,ln_w,we,wl,wr,wres
  // 90:out.b 91:out.w 92:proj.b 93:proj.w 94:edge_index 95:batch
  auto mabAt = [&](int b) {
    MabP m;
    m.bk = F(b + 0);  m.bo = F(b + 1);  m.bq = F(b + 2);  m.bv = F(b + 3);
    m.lin_b = F(b + 4); m.lin_w = F(b + 5);
    m.ln1_b = F(b + 6); m.ln1_w = F(b + 7);
    m.ln2_b = F(b + 8); m.ln2_w = F(b + 9);
    m.wk = F(b + 10); m.wo = F(b + 11); m.wq = F(b + 12); m.wv = F(b + 13);
    return m;
  };
  const float* x_in   = F(0);
  const float* eattr  = F(1);
  const int*   eidx   = (const int*)d_in[94];
  const int*   srcp   = eidx;
  const int*   dstp   = eidx + kE;
  const float* proj_b = F(92);
  const float* proj_w = F(93);
  const float* out_b  = F(90);
  const float* out_w  = F(91);

  // ---- workspace carve-out ----
  char* base = (char*)d_ws;
  size_t off = 0;
  auto alloc = [&](size_t nf) {
    float* p = (float*)(base + off);
    off += ((nf * sizeof(float) + 255) / 256) * 256;
    return p;
  };
  float* hbuf = alloc((size_t)kN * kHID);     // current node features
  float* hn   = alloc((size_t)kN * kHID);     // graph-LN output
  float* xl   = alloc((size_t)kN * kC2);      // also: hlin (GMT), h2 (pma2)
  float* xr   = alloc((size_t)kN * kC2);      // also: k buffer (GMT)
  float* agg  = alloc((size_t)kN * kC2);      // also: v buffer (GMT)
  float* hjk  = alloc((size_t)kN * kC2);      // JK concat [N,1024]
  float* ebuf = alloc((size_t)kE * kHEADS);   // logits -> exp
  float* lmax = alloc((size_t)kN * kHEADS);
  float* den  = alloc((size_t)kN * kHEADS);
  float* qbuf = alloc((size_t)256 * kC2);     // broadcast q_in
  float* qt   = alloc((size_t)256 * kC2);
  float* obuf = alloc((size_t)256 * kC2);
  float* tbuf = alloc((size_t)256 * kC2);
  float* z1   = alloc((size_t)256 * kC2);
  float* z2   = alloc((size_t)256 * kC2);
  float* z3   = alloc((size_t)16 * kC2);
  (void)ws_size; (void)n_in; (void)in_sizes; (void)out_size;

  // ================= input projection =================
  gemm(x_in, proj_w, proj_b, nullptr, hbuf, kN, kHID, kINC, 0, stream);

  // ================= GAT layers =================
  for (int i = 0; i < kL; ++i) {
    const int lb = 50 + 10 * i;
    const float* att  = F(lb + 0);
    const float* bias = F(lb + 1);
    const float* bl   = F(lb + 2);
    const float* br   = F(lb + 3);
    const float* ln_b = F(lb + 4);
    const float* ln_w = F(lb + 5);
    const float* we   = F(lb + 6);
    const float* wl   = F(lb + 7);
    const float* wr   = F(lb + 8);
    const float* wres = F(lb + 9);
    const int oc = (i < kL - 1) ? (kHID / kHEADS) : kHID;
    const int hc = kHEADS * oc;

    graph_ln_kernel<<<kB, 256, 0, stream>>>(hbuf, hn, ln_w, ln_b, kHID);
    gemm(hn, wl, bl, nullptr, xl, kN, hc, kHID, 0, stream);
    gemm(hn, wr, br, nullptr, xr, kN, hc, kHID, 0, stream);

    fill(lmax, -3.0e38f, (long long)kN * kHEADS, stream);
    fill(den, 0.0f, (long long)kN * kHEADS, stream);
    fill(agg, 0.0f, (long long)kN * hc, stream);

    edge_logit_kernel<<<(kE * kHEADS + 255) / 256, 256, 0, stream>>>(
        xl, xr, eattr, we, att, srcp, dstp, ebuf, lmax, oc);
    edge_exp_kernel<<<(kE * kHEADS + 255) / 256, 256, 0, stream>>>(
        ebuf, lmax, den, dstp);
    long long sn = (long long)kE * hc;
    edge_scatter_kernel<<<(unsigned)((sn + 255) / 256), 256, 0, stream>>>(
        ebuf, den, xl, agg, srcp, dstp, oc);

    const float* aggregated = agg;
    if (i == kL - 1) {  // mean over heads -> [N,256]; reuse xl as temp
      long long mn = (long long)kN * kHID;
      head_mean_kernel<<<(unsigned)((mn + 255) / 256), 256, 0, stream>>>(agg, xl, kHID, mn);
      aggregated = xl;
    }
    // h = aggregated + hn @ wres + bias
    gemm(hn, wres, bias, aggregated, hbuf, kN, kHID, kHID, 0, stream);
    // JK cat slice
    long long cn = (long long)kN * kHID;
    copy_slice_kernel<<<(unsigned)((cn + 255) / 256), 256, 0, stream>>>(
        hbuf, hjk, kHID, kC2, i * kHID, cn);
  }

  // ================= GMT =================
  // ---- PMA1: hlin = relu(hjk @ lin_w + lin_b); mab(seed_bcast, hlin) ----
  gemm(hjk, F(3), F(2), nullptr, xl, kN, kC2, kC2, 1, stream);          // hlin -> xl
  bcast_kernel<<<(256 * kC2 + 255) / 256, 256, 0, stream>>>(
      qbuf, F(18), (long long)16 * kC2, (long long)256 * kC2);          // seed -> [B,16,C]
  run_mab(qbuf, xl, z1, kB, 16, kNPG, mabAt(4), qt, xr, agg, obuf, tbuf, stream);

  // ---- SAB: mab(z1, z1) ----
  run_mab(z1, z1, z2, kB, 16, 16, mabAt(36), qt, xr, agg, obuf, tbuf, stream);

  // ---- PMA2: h2 = relu(z2 @ lin_w + lin_b); mab(seed2_bcast, h2) ----
  gemm(z2, F(20), F(19), nullptr, xl, 256, kC2, kC2, 1, stream);        // h2 -> xl
  bcast_kernel<<<(16 * kC2 + 255) / 256, 256, 0, stream>>>(
      qbuf, F(35), (long long)kC2, (long long)16 * kC2);                // seed2 -> [B,1,C]
  run_mab(qbuf, xl, z3, kB, 1, 16, mabAt(21), qt, xr, agg, obuf, tbuf, stream);

  // ---- output head: [16,1024] @ [1024,32] + b ----
  gemm(z3, out_w, out_b, nullptr, (float*)d_out, kB, kOUTC, kC2, 0, stream);
}
